// AbPred_both_4544075399802
// MI455X (gfx1250) — matrix operand order repaired and checked by reading, not compiled, not run on hardware
//
#include <hip/hip_runtime.h>

#define HDIM   1024
#define NBR    16
#define NOUT   4
#define NSEL   2048
#define TILE_M 32

typedef __attribute__((ext_vector_type(16))) __bf16 v16bf;
typedef __attribute__((ext_vector_type(8)))  __bf16 v8bf;
typedef __attribute__((ext_vector_type(8)))  float  v8f;

// A-fragment (16x32 bf16): per lane, 8 elems at K=k and 8 elems at K=k+16
__device__ __forceinline__ v16bf frag_gap16(const __bf16* p) {
  v8bf lo = *(const v8bf*)p;
  v8bf hi = *(const v8bf*)(p + 16);
  return __builtin_shufflevector(lo, hi, 0,1,2,3,4,5,6,7,8,9,10,11,12,13,14,15);
}
// B-fragment (32x16 bf16): per lane, 16 contiguous K values of one column
__device__ __forceinline__ v16bf frag_contig(const __bf16* p) {
  v8bf lo = *(const v8bf*)p;
  v8bf hi = *(const v8bf*)(p + 8);
  return __builtin_shufflevector(lo, hi, 0,1,2,3,4,5,6,7,8,9,10,11,12,13,14,15);
}

#define WMMA_BF16(a, b, c) \
  __builtin_amdgcn_wmma_f32_16x16x32_bf16(false, (a), false, (b), (short)0, (c), false, false)

// ---------------------------------------------------------------------------
// Kernel 1: fp32 [a][k][n] -> bf16 [a][n][k] (transposed, WMMA-B friendly)
// ---------------------------------------------------------------------------
__global__ __launch_bounds__(256) void transpose_convert(const float* __restrict__ W,
                                                         __bf16* __restrict__ Wt) {
  __shared__ float tile[32][33];
  const int a  = blockIdx.z;
  const int k0 = blockIdx.x * 32;
  const int n0 = blockIdx.y * 32;
  const int tx = threadIdx.x;   // 0..31
  const int ty = threadIdx.y;   // 0..7
  const float* src = W  + (size_t)a * HDIM * HDIM;
  __bf16*      dst = Wt + (size_t)a * HDIM * HDIM;
#pragma unroll
  for (int i = 0; i < 4; ++i)
    tile[ty + i * 8][tx] = src[(size_t)(k0 + ty + i * 8) * HDIM + n0 + tx];
  __syncthreads();
#pragma unroll
  for (int i = 0; i < 4; ++i)
    dst[(size_t)(n0 + ty + i * 8) * HDIM + k0 + tx] = (__bf16)tile[tx][ty + i * 8];
}

// ---------------------------------------------------------------------------
// Kernel 2: gather selected rows + convert to bf16 (compute only P rows!)
// ---------------------------------------------------------------------------
__global__ __launch_bounds__(256) void gather_convert(const float* __restrict__ X,
                                                      const float* __restrict__ X1,
                                                      const int* __restrict__ pos,
                                                      __bf16* __restrict__ Xg,
                                                      __bf16* __restrict__ X1g) {
  const int p   = blockIdx.x;
  const int src = pos[p];
  const float* xr  = X  + (size_t)src * HDIM;
  const float* x1r = X1 + (size_t)src * HDIM;
  for (int c = threadIdx.x; c < HDIM; c += 256) {
    Xg [(size_t)p * HDIM + c] = (__bf16)xr[c];
    X1g[(size_t)p * HDIM + c] = (__bf16)x1r[c];
  }
}

// ---------------------------------------------------------------------------
// Kernel 3: fused per-(row-tile, branch) pipeline:
//   z = X*W1 + X1*W2 + b -> h = relu(z*W3 + b3) -> LayerNorm -> y = hn*W4 + b4
// Each wave owns 128 output columns, processed as 4 pairs of 16-col subtiles
// so every A-fragment LDS read feeds two WMMAs (1 ds_load : 1 wmma ratio).
// ---------------------------------------------------------------------------
__global__ __launch_bounds__(256, 1) void fused_branch(
    const __bf16* __restrict__ Xg,  const __bf16* __restrict__ X1g,
    const __bf16* __restrict__ Wt1, const __bf16* __restrict__ Wt2,
    const __bf16* __restrict__ Wt3,
    const float* __restrict__ b1,   const float* __restrict__ b2,
    const float* __restrict__ b3,   const float* __restrict__ gamma,
    const float* __restrict__ beta, const float* __restrict__ W4,
    const float* __restrict__ b4,   float* __restrict__ out) {
  extern __shared__ char smem[];
  __bf16* sX  = (__bf16*)(smem);               //  [32][1024] bf16  (64KB)
  __bf16* sX1 = (__bf16*)(smem + 65536);       //  [32][1024] bf16  (64KB)
  __bf16* sZ  = (__bf16*)(smem + 131072);      //  [32][1024] bf16  (64KB)
  float*  sH  = (float*)(smem);                //  [32][1024] f32, overlaps sX/sX1
  float*  sMu = (float*)(smem + 196608);       //  [32]
  float*  sRs = (float*)(smem + 196608 + 128); //  [32]

  const int tileM = blockIdx.x;      // row tile (32 rows)
  const int a     = blockIdx.y;      // branch
  const int t     = threadIdx.x;
  const int wave  = t >> 5;          // 8 wave32s
  const int lane  = t & 31;
  const int lm    = lane & 15;
  const int lhalf = lane >> 4;

  // Stage gathered activations into LDS (16B vector copies)
  {
    const v8bf* gX  = (const v8bf*)(Xg  + (size_t)tileM * TILE_M * HDIM);
    const v8bf* gX1 = (const v8bf*)(X1g + (size_t)tileM * TILE_M * HDIM);
    v8bf* dX  = (v8bf*)sX;
    v8bf* dX1 = (v8bf*)sX1;
    for (int i = t; i < TILE_M * HDIM / 8; i += 256) {
      dX[i]  = gX[i];
      dX1[i] = gX1[i];
    }
  }
  __syncthreads();

  // ---- GEMM1: z = X*W1 + X1*W2 + (b1+b2) ----
  {
    const __bf16* Wa  = Wt1 + (size_t)a * HDIM * HDIM;  // [n][k]
    const __bf16* Wb  = Wt2 + (size_t)a * HDIM * HDIM;
    const float*  bb1 = b1 + a * HDIM;
    const float*  bb2 = b2 + a * HDIM;
    const __bf16* a0base  = sX  + (size_t)lm * HDIM + lhalf * 8;
    const __bf16* a1base  = sX  + (size_t)(16 + lm) * HDIM + lhalf * 8;
    const __bf16* a0bbase = sX1 + (size_t)lm * HDIM + lhalf * 8;
    const __bf16* a1bbase = sX1 + (size_t)(16 + lm) * HDIM + lhalf * 8;
    for (int sp = 0; sp < 4; ++sp) {
      const int nA = wave * 128 + sp * 32 + lm;   // subtile 0 column
      const int nB = nA + 16;                     // subtile 1 column
      const __bf16* colA0 = Wa + (size_t)nA * HDIM + lhalf * 8;
      const __bf16* colA1 = Wa + (size_t)nB * HDIM + lhalf * 8;
      const __bf16* colB0 = Wb + (size_t)nA * HDIM + lhalf * 8;
      const __bf16* colB1 = Wb + (size_t)nB * HDIM + lhalf * 8;
      v8f c00 = {0.f,0.f,0.f,0.f,0.f,0.f,0.f,0.f};  // subtile0, rows 0-15
      v8f c01 = {0.f,0.f,0.f,0.f,0.f,0.f,0.f,0.f};  // subtile0, rows 16-31
      v8f c10 = {0.f,0.f,0.f,0.f,0.f,0.f,0.f,0.f};  // subtile1, rows 0-15
      v8f c11 = {0.f,0.f,0.f,0.f,0.f,0.f,0.f,0.f};  // subtile1, rows 16-31
      for (int k0 = 0; k0 < HDIM; k0 += 32) {
        v16bf fa0  = frag_gap16(a0base + k0);
        v16bf fa1  = frag_gap16(a1base + k0);
        v16bf fb10 = frag_gap16(colA0 + k0);
        v16bf fb11 = frag_gap16(colA1 + k0);
        c00 = WMMA_BF16(fa0, fb10, c00);
        c01 = WMMA_BF16(fa1, fb10, c01);
        c10 = WMMA_BF16(fa0, fb11, c10);
        c11 = WMMA_BF16(fa1, fb11, c11);
        v16bf fa0b = frag_gap16(a0bbase + k0);
        v16bf fa1b = frag_gap16(a1bbase + k0);
        v16bf fb20 = frag_gap16(colB0 + k0);
        v16bf fb21 = frag_gap16(colB1 + k0);
        c00 = WMMA_BF16(fa0b, fb20, c00);
        c01 = WMMA_BF16(fa1b, fb20, c01);
        c10 = WMMA_BF16(fa0b, fb21, c10);
        c11 = WMMA_BF16(fa1b, fb21, c11);
      }
      const float zbA = bb1[nA] + bb2[nA];
      const float zbB = bb1[nB] + bb2[nB];
#pragma unroll
      for (int v = 0; v < 8; ++v) {
        const int r = v + 8 * lhalf;
        sZ[(size_t)r * HDIM + nA]        = (__bf16)(c00[v] + zbA);
        sZ[(size_t)(16 + r) * HDIM + nA] = (__bf16)(c01[v] + zbA);
        sZ[(size_t)r * HDIM + nB]        = (__bf16)(c10[v] + zbB);
        sZ[(size_t)(16 + r) * HDIM + nB] = (__bf16)(c11[v] + zbB);
      }
    }
  }
  __syncthreads();

  // ---- GEMM2: h = relu(z*W3 + b3), fp32 h into LDS (overlaps X tiles) ----
  {
    const __bf16* Wc  = Wt3 + (size_t)a * HDIM * HDIM;
    const float*  bb3 = b3 + a * HDIM;
    const __bf16* a0base = sZ + (size_t)lm * HDIM + lhalf * 8;
    const __bf16* a1base = sZ + (size_t)(16 + lm) * HDIM + lhalf * 8;
    for (int sp = 0; sp < 4; ++sp) {
      const int nA = wave * 128 + sp * 32 + lm;
      const int nB = nA + 16;
      const __bf16* colC0 = Wc + (size_t)nA * HDIM + lhalf * 8;
      const __bf16* colC1 = Wc + (size_t)nB * HDIM + lhalf * 8;
      v8f c00 = {0.f,0.f,0.f,0.f,0.f,0.f,0.f,0.f};
      v8f c01 = {0.f,0.f,0.f,0.f,0.f,0.f,0.f,0.f};
      v8f c10 = {0.f,0.f,0.f,0.f,0.f,0.f,0.f,0.f};
      v8f c11 = {0.f,0.f,0.f,0.f,0.f,0.f,0.f,0.f};
      for (int k0 = 0; k0 < HDIM; k0 += 32) {
        v16bf fa0 = frag_gap16(a0base + k0);
        v16bf fa1 = frag_gap16(a1base + k0);
        v16bf fb0 = frag_gap16(colC0 + k0);
        v16bf fb1 = frag_gap16(colC1 + k0);
        c00 = WMMA_BF16(fa0, fb0, c00);
        c01 = WMMA_BF16(fa1, fb0, c01);
        c10 = WMMA_BF16(fa0, fb1, c10);
        c11 = WMMA_BF16(fa1, fb1, c11);
      }
      const float hbA = bb3[nA];
      const float hbB = bb3[nB];
#pragma unroll
      for (int v = 0; v < 8; ++v) {
        const int r = v + 8 * lhalf;
        float h00 = c00[v] + hbA; h00 = h00 > 0.f ? h00 : 0.f;
        float h01 = c01[v] + hbA; h01 = h01 > 0.f ? h01 : 0.f;
        float h10 = c10[v] + hbB; h10 = h10 > 0.f ? h10 : 0.f;
        float h11 = c11[v] + hbB; h11 = h11 > 0.f ? h11 : 0.f;
        sH[(size_t)r * HDIM + nA]        = h00;
        sH[(size_t)(16 + r) * HDIM + nA] = h01;
        sH[(size_t)r * HDIM + nB]        = h10;
        sH[(size_t)(16 + r) * HDIM + nB] = h11;
      }
    }
  }
  __syncthreads();

  // ---- LayerNorm stats: each wave reduces 4 rows ----
  for (int r = wave * 4; r < wave * 4 + 4; ++r) {
    float s = 0.f, s2 = 0.f;
    for (int k = lane; k < HDIM; k += 32) {
      const float v = sH[(size_t)r * HDIM + k];
      s += v; s2 += v * v;
    }
#pragma unroll
    for (int off = 16; off > 0; off >>= 1) {
      s  += __shfl_xor(s,  off, 32);
      s2 += __shfl_xor(s2, off, 32);
    }
    if (lane == 0) {
      const float mu  = s * (1.0f / HDIM);
      const float var = s2 * (1.0f / HDIM) - mu * mu;
      sMu[r] = mu;
      sRs[r] = rsqrtf(var + 1e-5f);
    }
  }
  __syncthreads();

  // ---- GEMM3 (H x 4, tiny): LayerNorm affine fused; 2 threads per (r,o) ----
  {
    const int r    = t >> 3;         // 0..31
    const int o    = (t >> 1) & 3;   // 0..3
    const int half = t & 1;          // K split
    const float mu = sMu[r], rs = sRs[r];
    const float* g  = gamma + a * HDIM;
    const float* bt = beta  + a * HDIM;
    const float* w4 = W4 + (size_t)a * HDIM * NOUT + o;
    float acc = 0.f;
    for (int k = half * (HDIM / 2); k < (half + 1) * (HDIM / 2); ++k) {
      const float hn = (sH[(size_t)r * HDIM + k] - mu) * rs * g[k] + bt[k];
      acc += hn * w4[(size_t)k * NOUT];
    }
    acc += __shfl_xor(acc, 1, 32);   // combine the two K halves
    if (half == 0) {
      const int p = tileM * TILE_M + r;
      out[(size_t)p * (NBR * NOUT) + a * NOUT + o] = acc + b4[a * NOUT + o];
    }
  }
}

// ---------------------------------------------------------------------------
extern "C" void kernel_launch(void* const* d_in, const int* in_sizes, int n_in,
                              void* d_out, int out_size, void* d_ws, size_t ws_size,
                              hipStream_t stream) {
  (void)in_sizes; (void)n_in; (void)out_size; (void)ws_size;
  const float* X     = (const float*)d_in[0];
  const float* X1    = (const float*)d_in[1];
  const float* W1    = (const float*)d_in[2];
  const float* b1    = (const float*)d_in[3];
  const float* W2    = (const float*)d_in[4];
  const float* b2    = (const float*)d_in[5];
  const float* W3    = (const float*)d_in[6];
  const float* b3    = (const float*)d_in[7];
  const float* gamma = (const float*)d_in[8];
  const float* beta  = (const float*)d_in[9];
  const float* W4    = (const float*)d_in[10];
  const float* b4    = (const float*)d_in[11];
  const int*   pos   = (const int*)d_in[12];
  float* out = (float*)d_out;

  char* ws = (char*)d_ws;
  const size_t WSZ = (size_t)NBR * HDIM * HDIM * sizeof(__bf16);  // 32 MB
  __bf16* Wt1 = (__bf16*)(ws);
  __bf16* Wt2 = (__bf16*)(ws + WSZ);
  __bf16* Wt3 = (__bf16*)(ws + 2 * WSZ);
  __bf16* Xg  = (__bf16*)(ws + 3 * WSZ);
  __bf16* X1g = (__bf16*)(ws + 3 * WSZ + (size_t)NSEL * HDIM * sizeof(__bf16));

  // 1) weights fp32 -> bf16, transposed [a][n][k]
  dim3 tb(32, 8);
  dim3 tg(HDIM / 32, HDIM / 32, NBR);
  transpose_convert<<<tg, tb, 0, stream>>>(W1, Wt1);
  transpose_convert<<<tg, tb, 0, stream>>>(W2, Wt2);
  transpose_convert<<<tg, tb, 0, stream>>>(W3, Wt3);

  // 2) gather selected rows (compute only P of B rows), convert to bf16
  gather_convert<<<NSEL, 256, 0, stream>>>(X, X1, pos, Xg, X1g);

  // 3) fused WMMA pipeline: 64 row-tiles x 16 branches, 8 wave32 per WG
  const size_t smem_bytes = 196608 + 256;  // 3x64KB tiles + stats
  fused_branch<<<dim3(NSEL / TILE_M, NBR), 256, smem_bytes, stream>>>(
      Xg, X1g, Wt1, Wt2, Wt3, b1, b2, b3, gamma, beta, W4, b4, out);
}
